// Actor_46497315947046
// MI455X (gfx1250) — matrix operand order repaired and checked by reading, not compiled, not run on hardware
//
#include <hip/hip_runtime.h>

#define BATCH   16384
#define OBS     512
#define ACT     256
#define NEXP    16
#define TOPK    4

typedef __attribute__((ext_vector_type(16))) __bf16 v16bf;
typedef __attribute__((ext_vector_type(8)))  float  v8f;

union Frag {
    uint4 u[2];
    v16bf v;
};

// round-to-nearest-even f32 -> bf16, packed pair into one uint (lo = a, hi = b)
__device__ __forceinline__ unsigned pack_bf16(float a, float b) {
    unsigned ua = __float_as_uint(a);
    unsigned ub = __float_as_uint(b);
    unsigned ra = (ua + 0x7FFFu + ((ua >> 16) & 1u)) >> 16;
    unsigned rb = (ub + 0x7FFFu + ((ub >> 16) & 1u)) >> 16;
    return (rb << 16) | (ra & 0xFFFFu);
}

// ---------------------------------------------------------------- zero counters
__global__ void zcnt_k(unsigned* cnt) {
    if (threadIdx.x < NEXP) cnt[threadIdx.x] = 0u;
}

// ---------------------------------------------------------------- f32 -> bf16 weight conversion
__global__ void cvt_k(const float* __restrict__ src, unsigned* __restrict__ dst, int npairs) {
    int i = blockIdx.x * blockDim.x + threadIdx.x;
    if (i < npairs) dst[i] = pack_bf16(src[2 * i], src[2 * i + 1]);
}

// ---------------------------------------------------------------- router: logits, softmax, top-4, scatter
__global__ __launch_bounds__(256) void router_k(
    const float* __restrict__ x, const float* __restrict__ rw,
    const float* __restrict__ rb, float* __restrict__ w_full,
    unsigned* __restrict__ rowid, unsigned* __restrict__ cnt)
{
    const int wv   = threadIdx.x >> 5;
    const int lane = threadIdx.x & 31;
    const int row  = blockIdx.x * 8 + wv;

    const float4* x4 = (const float4*)(x + (size_t)row * OBS);
    const float4* w4 = (const float4*)rw;

    float acc[NEXP];
#pragma unroll
    for (int e = 0; e < NEXP; ++e) acc[e] = 0.f;

#pragma unroll
    for (int i = 0; i < 4; ++i) {
        float4 xv = x4[i * 32 + lane];
#pragma unroll
        for (int e = 0; e < NEXP; ++e) {
            float4 wvv = w4[e * 128 + i * 32 + lane];
            acc[e] += xv.x * wvv.x + xv.y * wvv.y + xv.z * wvv.z + xv.w * wvv.w;
        }
    }
#pragma unroll
    for (int e = 0; e < NEXP; ++e) {
#pragma unroll
        for (int off = 16; off > 0; off >>= 1)
            acc[e] += __shfl_xor(acc[e], off, 32);
        acc[e] += rb[e];
    }
    // all lanes now hold identical logits
    float mx = acc[0];
#pragma unroll
    for (int e = 1; e < NEXP; ++e) mx = fmaxf(mx, acc[e]);
    float sum = 0.f, myLogit = 0.f;
#pragma unroll
    for (int e = 0; e < NEXP; ++e) {
        sum += expf(acc[e] - mx);
        if (e == lane) myLogit = acc[e];
    }
    unsigned sel = 0u;
    int myE = 0;
    for (int k = 0; k < TOPK; ++k) {           // ties -> lowest index (matches lax.top_k)
        float best = -INFINITY;
        int bi = 0;
#pragma unroll
        for (int e = 0; e < NEXP; ++e) {
            bool take = (((sel >> e) & 1u) == 0u) && (acc[e] > best);
            best = take ? acc[e] : best;
            bi   = take ? e : bi;
        }
        sel |= 1u << bi;
        if (lane == k) myE = bi;
    }
    float p = expf(myLogit - mx) / sum;
    if (lane < NEXP)
        w_full[(size_t)row * NEXP + lane] = ((sel >> lane) & 1u) ? p : 0.f;
    if (lane < TOPK) {
        unsigned slot = atomicAdd(&cnt[myE], 1u);
        rowid[(size_t)myE * BATCH + slot] = (unsigned)row | ((unsigned)lane << 14);
    }
}

// ---------------------------------------------------------------- gathered expert GEMM (bf16 WMMA)
// block = 256 thr (8 waves), handles 64 rows of one expert segment, full 2x256 outputs.
// wave w: head = w&1, col group = (w>>1)*64 .. +63 ; 4 col-tiles x 4 row-tiles of 16x16.
__global__ __launch_bounds__(256) void gemm_k(
    const float* __restrict__ x, const unsigned short* __restrict__ wb16,
    const float* __restrict__ w_full, const unsigned* __restrict__ rowid,
    const unsigned* __restrict__ cnt, float* __restrict__ accPart)
{
    __shared__ unsigned short xt[64 * OBS];      // 64 KB, bf16, XOR-swizzled 16B chunks

    const int e = blockIdx.y;
    const int n = (int)cnt[e];
    const int tileStart = blockIdx.x * 64;
    if (tileStart >= n) return;

    const int tid = threadIdx.x;

    // ---- stage x-tile into LDS as p-scaled bf16 (64 rows x 512 cols) ----
    for (int it = 0; it < 16; ++it) {
        int g = it * 256 + tid;                  // 0..4095 : 16B chunk id
        int r = g >> 6;                          // row in tile
        int c = g & 63;                          // chunk within row (8 bf16)
        int j = tileStart + r;
        float4 a0 = make_float4(0.f, 0.f, 0.f, 0.f);
        float4 a1 = a0;
        float p = 0.f;
        if (j < n) {
            unsigned ent = rowid[(size_t)e * BATCH + j];
            int b = (int)(ent & 0x3FFFu);
            p = w_full[(size_t)b * NEXP + e];
            const float4* xp = (const float4*)(x + (size_t)b * OBS) + c * 2;
            a0 = xp[0];
            a1 = xp[1];
        }
        uint4 pk;
        pk.x = pack_bf16(a0.x * p, a0.y * p);
        pk.y = pack_bf16(a0.z * p, a0.w * p);
        pk.z = pack_bf16(a1.x * p, a1.y * p);
        pk.w = pack_bf16(a1.z * p, a1.w * p);
        int cc = c ^ (r & 15);                   // bank-conflict-avoiding swizzle
        *(uint4*)((char*)xt + (size_t)r * 1024 + (size_t)cc * 16) = pk;
    }
    __syncthreads();

    const int wv      = tid >> 5;
    const int lane    = tid & 31;
    const int halfsel = (lane >> 4) & 1;
    const int l15     = lane & 15;
    const int h       = wv & 1;                  // 0 = mean, 1 = log_std
    const int cg      = wv >> 1;                 // col group: 64 cols

    const unsigned short* wbase = wb16 + ((size_t)(h * NEXP + e) * ACT) * OBS;

    v8f acc[4][4] = {};                          // [col-tile][row-tile]

    for (int K = 0; K < OBS; K += 32) {
        // A fragments: lane holds row (l15), K chunks {+0,+16} (lanes<16) / {+8,+24} (lanes>=16)
        Frag A[4];
        int c0 = (K >> 3) + halfsel;
#pragma unroll
        for (int rt = 0; rt < 4; ++rt) {
            int r = rt * 16 + l15;
            const char* base = (const char*)xt + (size_t)r * 1024;
            A[rt].u[0] = *(const uint4*)(base + (size_t)((c0     ) ^ l15) * 16);
            A[rt].u[1] = *(const uint4*)(base + (size_t)((c0 + 2) ^ l15) * 16);
        }
#pragma unroll
        for (int ct = 0; ct < 4; ++ct) {
            // B fragment: lane holds col a, K chunks {+0,+16} (lanes<16) / {+8,+24} (lanes>=16)
            int a = cg * 64 + ct * 16 + l15;
            const unsigned short* wr = wbase + (size_t)a * OBS + K + (halfsel << 3);
            Frag B;
            B.u[0] = *(const uint4*)(wr);
            B.u[1] = *(const uint4*)(wr + 16);
#pragma unroll
            for (int rt = 0; rt < 4; ++rt)
                acc[ct][rt] = __builtin_amdgcn_wmma_f32_16x16x32_bf16(
                    false, A[rt].v, false, B.v, (short)0, acc[ct][rt], false, false);
        }
    }

    // ---- epilogue: deterministic scatter into per-(b,k) partial buffer ----
#pragma unroll
    for (int rt = 0; rt < 4; ++rt) {
        int bk[8];
#pragma unroll
        for (int r = 0; r < 8; ++r) {
            int m = rt * 16 + halfsel * 8 + r;
            int j = tileStart + m;
            bk[r] = (j < n) ? (int)rowid[(size_t)e * BATCH + j] : -1;
        }
#pragma unroll
        for (int ct = 0; ct < 4; ++ct) {
            int a = cg * 64 + ct * 16 + l15;
#pragma unroll
            for (int r = 0; r < 8; ++r) {
                if (bk[r] >= 0) {
                    int b = bk[r] & 0x3FFF;
                    int k = bk[r] >> 14;
                    accPart[((size_t)h * BATCH * TOPK + (size_t)b * TOPK + k) * ACT + a] =
                        acc[ct][rt][r];
                }
            }
        }
    }
}

// ---------------------------------------------------------------- finalize: partial sum + bias + tanh transform
__global__ __launch_bounds__(256) void final_k(
    const float* __restrict__ accPart, const float* __restrict__ w_full,
    const float* __restrict__ mean_b, const float* __restrict__ log_std_b,
    float* __restrict__ out)
{
    const int b = blockIdx.x;
    const int a = threadIdx.x;
    const float* pm = accPart + (size_t)b * TOPK * ACT + a;
    const float* pl = pm + (size_t)BATCH * TOPK * ACT;
    float m = pm[0] + pm[ACT] + pm[2 * ACT] + pm[3 * ACT];
    float l = pl[0] + pl[ACT] + pl[2 * ACT] + pl[3 * ACT];
#pragma unroll
    for (int e = 0; e < NEXP; ++e) {
        float w = w_full[(size_t)b * NEXP + e];
        m += w * mean_b[e * ACT + a];
        l += w * log_std_b[e * ACT + a];
    }
    l = tanhf(l);
    l = -5.0f + 3.5f * (l + 1.0f);               // LOG_STD_MIN + 0.5*(MAX-MIN)*(l+1)
    out[(size_t)b * ACT + a] = m;
    out[(size_t)BATCH * ACT + (size_t)b * ACT + a] = l;
}

// ---------------------------------------------------------------- launcher
extern "C" void kernel_launch(void* const* d_in, const int* in_sizes, int n_in,
                              void* d_out, int out_size, void* d_ws, size_t ws_size,
                              hipStream_t stream) {
    const float* x          = (const float*)d_in[0];
    const float* router_w   = (const float*)d_in[1];
    const float* router_b   = (const float*)d_in[2];
    const float* mean_w     = (const float*)d_in[3];
    const float* mean_b     = (const float*)d_in[4];
    const float* log_std_w  = (const float*)d_in[5];
    const float* log_std_b  = (const float*)d_in[6];
    (void)in_sizes; (void)n_in; (void)out_size; (void)ws_size;

    char* ws = (char*)d_ws;
    float* accPart = (float*)ws;                                   // 2*B*4*256 f32 = 128 MB
    size_t off = (size_t)2 * BATCH * TOPK * ACT * sizeof(float);
    unsigned short* wb16 = (unsigned short*)(ws + off);            // 2 heads bf16 weights = 8 MB
    off += (size_t)2 * NEXP * ACT * OBS * sizeof(unsigned short);
    float* w_full = (float*)(ws + off);                            // B x 16 sparse probs = 1 MB
    off += (size_t)BATCH * NEXP * sizeof(float);
    unsigned* rowid = (unsigned*)(ws + off);                       // 16 segments x B = 1 MB
    off += (size_t)NEXP * BATCH * sizeof(unsigned);
    unsigned* cnt = (unsigned*)(ws + off);                         // 16 counters

    zcnt_k<<<1, 32, 0, stream>>>(cnt);

    const int npairs = NEXP * ACT * OBS / 2;                       // 1,048,576 per head
    cvt_k<<<npairs / 256, 256, 0, stream>>>(mean_w, (unsigned*)wb16, npairs);
    cvt_k<<<npairs / 256, 256, 0, stream>>>(log_std_w, (unsigned*)wb16 + npairs, npairs);

    router_k<<<BATCH / 8, 256, 0, stream>>>(x, router_w, router_b, w_full, rowid, cnt);

    gemm_k<<<dim3(BATCH / 64, NEXP), 256, 0, stream>>>(x, wb16, w_full, rowid, cnt, accPart);

    final_k<<<BATCH, 256, 0, stream>>>(accPart, w_full, mean_b, log_std_b, (float*)d_out);
}
